// SOGAT_37486474560100
// MI455X (gfx1250) — matrix operand order repaired and checked by reading, not compiled, not run on hardware
//
#include <hip/hip_runtime.h>
#include <hip/hip_bf16.h>

typedef _Float16 f16;
typedef __attribute__((ext_vector_type(16))) _Float16 v16h;
typedef __attribute__((ext_vector_type(8)))  _Float16 v8h;
typedef __attribute__((ext_vector_type(8)))  float    v8f;

#define BB    512
#define NN    62
#define CC    8
#define TT    64
#define KTOP  8
#define HEADS 4
#define OUTD  128
#define DIN   512                 // C*T == HEADS*OUT
#define VNODE (BB*NN)             // 31744 = 64 * 496

// ---------------------------------------------------------------- f32 -> f16
__global__ void cvt_f16_kernel(const float* __restrict__ in, f16* __restrict__ out, int n) {
    int i = blockIdx.x * blockDim.x + threadIdx.x;
    if (i < n) out[i] = (f16)in[i];
}

// ---------------------------------------------- prod[b,n,m] = (x·W1)·W2
__global__ void spatial_prod_kernel(const float* __restrict__ x, const float* __restrict__ W1,
                                    const float* __restrict__ W2, float* __restrict__ prod) {
    int bn = blockIdx.x;                       // b*NN + n
    __shared__ float y[CC];
    __shared__ float w1s[TT];
    int t = threadIdx.x;                       // 64 threads
    w1s[t] = W1[t];
    __syncthreads();
    if (t < CC) {
        const float* xp = x + ((size_t)bn * CC + t) * TT;
        float s = 0.f;
        for (int k = 0; k < TT; ++k) s += xp[k] * w1s[k];
        y[t] = s;
    }
    __syncthreads();
    if (t < NN) {
        float s = 0.f;
        for (int c = 0; c < CC; ++c) s += y[c] * W2[c * NN + t];
        prod[(size_t)bn * NN + t] = s;
    }
}

// ---------------------------------------------- S = tanh(prod @ Vsp + bsp)
__global__ void spatial_S_kernel(const float* __restrict__ prod, const float* __restrict__ Vsp,
                                 const float* __restrict__ bsp, float* __restrict__ S) {
    int bn = blockIdx.x;
    int n  = bn % NN;
    __shared__ float pr[NN];
    int t = threadIdx.x;                       // 64 threads
    if (t < NN) pr[t] = prod[(size_t)bn * NN + t];
    __syncthreads();
    if (t < NN) {
        float s = 0.f;
        for (int k = 0; k < NN; ++k) s += pr[k] * Vsp[k * NN + t];
        S[(size_t)bn * NN + t] = tanhf(s + bsp[n * NN + t]);
    }
}

// ---------------------------------------------- BatchNorm stats per channel n
__global__ void bn_stats_kernel(const float* __restrict__ S, float* __restrict__ mean,
                                float* __restrict__ rstd) {
    int n = blockIdx.x;
    int t = threadIdx.x;                       // 256 threads
    float s = 0.f, s2 = 0.f;
    for (int i = t; i < BB * NN; i += 256) {
        int b = i / NN, m = i % NN;
        float v = S[((size_t)b * NN + n) * NN + m];
        s += v; s2 += v * v;
    }
    __shared__ float rs[256], rs2[256];
    rs[t] = s; rs2[t] = s2;
    __syncthreads();
    for (int o = 128; o > 0; o >>= 1) {
        if (t < o) { rs[t] += rs[t + o]; rs2[t] += rs2[t + o]; }
        __syncthreads();
    }
    if (t == 0) {
        float m  = rs[0] / (float)(BB * NN);
        float vr = rs2[0] / (float)(BB * NN) - m * m;
        mean[n] = m;
        rstd[n] = rsqrtf(vr + 1e-5f);
    }
}

// ---------------------------------------------- top-k per row -> dense mask[b,src,dst]
__global__ void topk_mask_kernel(const float* __restrict__ S, const float* __restrict__ mean,
                                 const float* __restrict__ rstd, const float* __restrict__ bnw,
                                 const float* __restrict__ bnb, float* __restrict__ mask) {
    int bn = blockIdx.x * blockDim.x + threadIdx.x;    // b*NN + row
    if (bn >= VNODE) return;
    int n = bn % NN;
    float mu = mean[n], rs = rstd[n], w = bnw[n], bb = bnb[n];
    const float* Sp = S + (size_t)bn * NN;
    float vals[NN], mrow[NN];
    for (int m = 0; m < NN; ++m) { vals[m] = (Sp[m] - mu) * rs * w + bb; mrow[m] = 0.f; }
    for (int k = 0; k < KTOP; ++k) {
        int best = 0; float bv = -__builtin_inff();
        for (int m = 0; m < NN; ++m) if (vals[m] > bv) { bv = vals[m]; best = m; }
        vals[best] = -__builtin_inff();
        if (best != n) mrow[best] = 1.f;       // remove_self_loops on top-k edges
    }
    mrow[n] = 1.f;                             // add_self_loops
    float* mp = mask + (size_t)bn * NN;
    for (int m = 0; m < NN; ++m) mp[m] = mrow[m];
}

// ------------------------------------------------------------------------------------------
// xl = A[V,512] @ W[512,512]^T via WMMA, register-blocked: each wave computes a 64x32 output
// block (4 M-subtiles x 2 N-subtiles) so A fragments are reused 2x and B fragments 4x:
// 12 b128-loads per 8 WMMAs instead of 4 loads per WMMA -> ~2.7x higher FLOP/byte from L2.
// ------------------------------------------------------------------------------------------
__global__ __launch_bounds__(128) void gemm_wmma_kernel(const f16* __restrict__ A,
                                                        const f16* __restrict__ W,
                                                        float* __restrict__ D) {
    int wave = (blockIdx.x * blockDim.x + threadIdx.x) >> 5;
    int lane = threadIdx.x & 31;
    const int NSUP = DIN / 32;                 // 16 supertile columns
    int nt = wave % NSUP;                      // supertile col (32 wide)
    int mt = wave / NSUP;                      // supertile row (64 tall), 0..495
    int hs = lane >> 4;                        // half-wave select
    int lr = lane & 15;

    // A fragment (16x32 f16): lane holds row,
    //   halves 0..15 -> K = k0 + hs*16 + (0..15)      (one contiguous 32B run)
    const f16* Ap[4];
    #pragma unroll
    for (int ms = 0; ms < 4; ++ms)
        Ap[ms] = A + (size_t)(mt * 64 + ms * 16 + lr) * DIN + hs * 16;
    // B fragment (32x16 f16): B[k][n] = W[n][k]; lane holds col,
    //   halves 0..15 -> K = k0 + hs*16 + (0..15)      (one contiguous 32B run)
    const f16* Wp[2];
    #pragma unroll
    for (int ns = 0; ns < 2; ++ns)
        Wp[ns] = W + (size_t)(nt * 32 + ns * 16 + lr) * DIN + hs * 16;

    v8f acc[4][2];
    #pragma unroll
    for (int ms = 0; ms < 4; ++ms)
        #pragma unroll
        for (int ns = 0; ns < 2; ++ns) acc[ms][ns] = (v8f){};

    #pragma unroll 2
    for (int kk = 0; kk < 16; ++kk) {
        int k0 = kk * 32;
        v16h a[4], b[2];
        #pragma unroll
        for (int ms = 0; ms < 4; ++ms) {
            v8h a0 = *(const v8h*)(Ap[ms] + k0);
            v8h a1 = *(const v8h*)(Ap[ms] + k0 + 8);
            a[ms] = __builtin_shufflevector(a0, a1, 0,1,2,3,4,5,6,7,8,9,10,11,12,13,14,15);
        }
        #pragma unroll
        for (int ns = 0; ns < 2; ++ns) b[ns] = *(const v16h*)(Wp[ns] + k0);
        #pragma unroll
        for (int ms = 0; ms < 4; ++ms)
            #pragma unroll
            for (int ns = 0; ns < 2; ++ns)
                acc[ms][ns] = __builtin_amdgcn_wmma_f32_16x16x32_f16(
                    false, a[ms], false, b[ns], (short)0, acc[ms][ns], false, false);
    }

    // D layout per subtile: lane 0-15 -> N=lr, M=+r ; lane 16-31 -> N=lr, M=8+r
    #pragma unroll
    for (int ms = 0; ms < 4; ++ms) {
        #pragma unroll
        for (int ns = 0; ns < 2; ++ns) {
            float* Dp = D + (size_t)(mt * 64 + ms * 16 + hs * 8) * DIN + nt * 32 + ns * 16 + lr;
            #pragma unroll
            for (int r = 0; r < 8; ++r) Dp[(size_t)r * DIN] = acc[ms][ns][r];
        }
    }
}

// ---------------------------------------------- attention logits al_src/al_dst [V,H]
__global__ void al_kernel(const float* __restrict__ xl, const float* __restrict__ asrc,
                          const float* __restrict__ adst, float* __restrict__ al_src,
                          float* __restrict__ al_dst) {
    int i = blockIdx.x * blockDim.x + threadIdx.x;     // v*HEADS + h
    if (i >= VNODE * HEADS) return;
    int v = i >> 2, h = i & 3;
    const float* xp = xl + (size_t)v * DIN + h * OUTD;
    const float* as = asrc + h * OUTD;
    const float* ad = adst + h * OUTD;
    float ss = 0.f, sd = 0.f;
    for (int c = 0; c < OUTD; ++c) { float xv = xp[c]; ss += xv * as[c]; sd += xv * ad[c]; }
    al_src[i] = ss; al_dst[i] = sd;
}

// -------------------- masked per-dst softmax + aggregation (+bias, relu) --------------------
// CONCAT layer writes f16 directly (it is only consumed by the f16 WMMA GEMM of layer 2),
// saving a full f32 round-trip + convert pass. Final layer writes f32 to d_out.
template <bool CONCAT, typename OutT>
__global__ __launch_bounds__(128) void agg_kernel(const float* __restrict__ xl,
                                                  const float* __restrict__ al_src,
                                                  const float* __restrict__ al_dst,
                                                  const float* __restrict__ mask,
                                                  const float* __restrict__ bias,
                                                  OutT* __restrict__ out) {
    int vd = blockIdx.x;                       // dst node b*NN + c
    int b = vd / NN, c = vd % NN;
    int t = threadIdx.x;
    __shared__ float wgt[NN][HEADS];
    __shared__ float dvals[HEADS];
    if (t < HEADS) dvals[t] = al_dst[vd * HEADS + t];
    __syncthreads();
    for (int i = t; i < NN * HEADS; i += 128) {
        int r = i / HEADS, h = i % HEADS;
        float m = mask[((size_t)b * NN + r) * NN + c];
        float a = -__builtin_inff();
        if (m > 0.5f) {
            float v = al_src[(b * NN + r) * HEADS + h] + dvals[h];
            a = v > 0.f ? v : 0.2f * v;        // leaky_relu(0.2)
        }
        wgt[r][h] = a;
    }
    __syncthreads();
    if (t < HEADS) {                           // per-head masked softmax over 62 sources
        float amax = -__builtin_inff();
        for (int r = 0; r < NN; ++r) amax = fmaxf(amax, wgt[r][t]);
        float denom = 0.f;
        for (int r = 0; r < NN; ++r) {
            float a = wgt[r][t];
            float e = (a == -__builtin_inff()) ? 0.f : expf(a - amax);
            wgt[r][t] = e; denom += e;
        }
        float inv = 1.f / denom;               // denom>0: self-loop always valid
        for (int r = 0; r < NN; ++r) wgt[r][t] *= inv;
    }
    __syncthreads();
    if (CONCAT) {
        for (int col = t; col < DIN; col += 128) {
            int h = col >> 7;
            const float* xb = xl + (size_t)b * NN * DIN + col;
            float acc = 0.f;
            for (int r = 0; r < NN; ++r) acc += wgt[r][h] * xb[(size_t)r * DIN];
            float o = acc + bias[col];
            out[(size_t)vd * DIN + col] = (OutT)(o > 0.f ? o : 0.f);
        }
    } else {
        int col = t;                           // 128 threads == OUTD
        const float* xb = xl + (size_t)b * NN * DIN;
        float acc = 0.f;
        for (int r = 0; r < NN; ++r) {
            const float* xr = xb + (size_t)r * DIN;
            float s = 0.f;
            for (int h = 0; h < HEADS; ++h) s += wgt[r][h] * xr[h * OUTD + col];
            acc += s;
        }
        float o = acc * 0.25f + bias[col];     // mean over heads
        out[(size_t)vd * OUTD + col] = (OutT)(o > 0.f ? o : 0.f);
    }
}

extern "C" void kernel_launch(void* const* d_in, const int* in_sizes, int n_in,
                              void* d_out, int out_size, void* d_ws, size_t ws_size,
                              hipStream_t stream) {
    (void)in_sizes; (void)n_in; (void)out_size; (void)ws_size;
    const float* x    = (const float*)d_in[0];
    /* d_in[1] edge_index: unused (spatial=True rebuilds graph) */
    const float* W1   = (const float*)d_in[2];
    const float* W2   = (const float*)d_in[3];
    const float* bsp  = (const float*)d_in[4];
    const float* Vsp  = (const float*)d_in[5];
    const float* bnw  = (const float*)d_in[6];
    const float* bnb  = (const float*)d_in[7];
    const float* g1w  = (const float*)d_in[8];
    const float* g1as = (const float*)d_in[9];
    const float* g1ad = (const float*)d_in[10];
    const float* g1b  = (const float*)d_in[11];
    const float* g2w  = (const float*)d_in[12];
    const float* g2as = (const float*)d_in[13];
    const float* g2ad = (const float*)d_in[14];
    const float* g2b  = (const float*)d_in[15];
    float* out = (float*)d_out;

    char* ws = (char*)d_ws;
    size_t off = 0;
    auto alloc = [&](size_t bytes) { size_t o = off; off += (bytes + 255) & ~(size_t)255; return o; };
    f16*   hf16  = (f16*)  (ws + alloc((size_t)VNODE * DIN * 2));
    f16*   w1f   = (f16*)  (ws + alloc((size_t)DIN * DIN * 2));
    f16*   w2f   = (f16*)  (ws + alloc((size_t)DIN * DIN * 2));
    float* prod  = (float*)(ws + alloc((size_t)VNODE * NN * 4));
    float* Sbuf  = (float*)(ws + alloc((size_t)VNODE * NN * 4));
    float* maskb = (float*)(ws + alloc((size_t)VNODE * NN * 4));
    float* meanb = (float*)(ws + alloc(256));
    float* rstdb = (float*)(ws + alloc(256));
    float* xl    = (float*)(ws + alloc((size_t)VNODE * DIN * 4));
    float* alS   = (float*)(ws + alloc((size_t)VNODE * HEADS * 4));
    float* alD   = (float*)(ws + alloc((size_t)VNODE * HEADS * 4));

    const int nAct = VNODE * DIN;              // 16,252,928
    const int nW   = DIN * DIN;                // 262,144
    const int gemmBlocks = (VNODE / 64) * (DIN / 32) / 4;   // 7936 waves / 4 per block = 1984

    // graph construction
    cvt_f16_kernel<<<(nAct + 255) / 256, 256, 0, stream>>>(x, hf16, nAct);
    cvt_f16_kernel<<<(nW + 255) / 256, 256, 0, stream>>>(g1w, w1f, nW);
    cvt_f16_kernel<<<(nW + 255) / 256, 256, 0, stream>>>(g2w, w2f, nW);
    spatial_prod_kernel<<<VNODE, 64, 0, stream>>>(x, W1, W2, prod);
    spatial_S_kernel<<<VNODE, 64, 0, stream>>>(prod, Vsp, bsp, Sbuf);
    bn_stats_kernel<<<NN, 256, 0, stream>>>(Sbuf, meanb, rstdb);
    topk_mask_kernel<<<(VNODE + 127) / 128, 128, 0, stream>>>(Sbuf, meanb, rstdb, bnw, bnb, maskb);

    // GAT layer 1 (concat heads) -> writes f16 activations for layer 2 directly
    gemm_wmma_kernel<<<gemmBlocks, 128, 0, stream>>>(hf16, w1f, xl);
    al_kernel<<<(VNODE * HEADS + 255) / 256, 256, 0, stream>>>(xl, g1as, g1ad, alS, alD);
    agg_kernel<true, f16><<<VNODE, 128, 0, stream>>>(xl, alS, alD, maskb, g1b, hf16);

    // GAT layer 2 (mean heads)
    gemm_wmma_kernel<<<gemmBlocks, 128, 0, stream>>>(hf16, w2f, xl);
    al_kernel<<<(VNODE * HEADS + 255) / 256, 256, 0, stream>>>(xl, g2as, g2ad, alS, alD);
    agg_kernel<false, float><<<VNODE, 128, 0, stream>>>(xl, alS, alD, maskb, g2b, out);
}
